// ComplexEqModel_70102456206001
// MI455X (gfx1250) — matrix operand order repaired and checked by reading, not compiled, not run on hardware
//
#include <hip/hip_runtime.h>
#include <hip/hip_bf16.h>

// ---------------- model constants ----------------
#define BB   4
#define TSEQ 1024
#define DM   896
#define NH   8
#define HSZ  112
#define DFF  3584
#define BT   4096                      // BB*TSEQ
#define BTD  ((size_t)BT * DM)         // 3,670,016 elements per [B,T,D] buffer
#define EPSV 1e-4f

typedef __attribute__((ext_vector_type(16))) _Float16 v16h;
typedef __attribute__((ext_vector_type(8)))  _Float16 v8h;
typedef __attribute__((ext_vector_type(8)))  float    v8f;
typedef __attribute__((ext_vector_type(4)))  float    v4f;
typedef __attribute__((ext_vector_type(4)))  unsigned int u32x4;
typedef __attribute__((ext_vector_type(8)))  int       i32x8;
typedef __attribute__((ext_vector_type(4)))  int       i32x4;

union V16 { v16h v; v8h h[2]; _Float16 f[16]; };
union H2  { _Float16 h[2]; unsigned int u; };

#if defined(__has_builtin)
# if __has_builtin(__builtin_amdgcn_tensor_load_to_lds) && __has_builtin(__builtin_amdgcn_s_wait_tensorcnt)
#  define HAVE_TDM 1
# endif
#endif
#ifndef HAVE_TDM
# define HAVE_TDM 0
#endif

static __device__ __forceinline__ v8f wmma_f16(v16h a, v16h b, v8f c) {
  // D = A(16x32 f16) * B(32x16 f16) + C(16x16 f32)
  return __builtin_amdgcn_wmma_f32_16x16x32_f16(false, a, false, b, (short)0, c,
                                                false, false);
}
// raw hardware sqrt/rcp (single v_sqrt_f32 / v_rcp_f32, no IEEE refinement)
static __device__ __forceinline__ float fast_sqrt(float x) { return __builtin_amdgcn_sqrtf(x); }
static __device__ __forceinline__ float fast_rcp(float x)  { return __builtin_amdgcn_rcpf(x); }

#if HAVE_TDM
// Issue one TDM load: 128 rows x 32 cols of f16 from [.., lda] row-major global
// into a packed 128x32 f16 LDS tile. D# fields per CDNA5 ISA sec 8.3/8.4.
static __device__ __forceinline__ void tdm_load_a(const _Float16* gsrc,
                                                  unsigned ldsByteAddr, int lda) {
  unsigned long long ga = (unsigned long long)gsrc;
  u32x4 g0;
  g0[0] = 1u;                                          // count=1, user mode
  g0[1] = ldsByteAddr;                                 // lds_addr (bytes)
  g0[2] = (unsigned)(ga & 0xFFFFFFFFu);                // global_addr[31:0]
  g0[3] = (unsigned)((ga >> 32) & 0x01FFFFFFu) | (2u << 30);  // addr[56:32]|type=2
  const unsigned td = 0x100000u;                       // huge tensor dims: no OOB clamp
  i32x8 g1;
  g1[0] = 0x00010000;                                  // data_size=1 (2 bytes), mask=0
  g1[1] = (int)((td & 0xFFFFu) << 16);                 // tensor_dim0[15:0]
  g1[2] = (int)(((td >> 16) & 0xFFFFu) | ((td & 0xFFFFu) << 16));  // dim0 hi | dim1 lo
  g1[3] = (int)(((td >> 16) & 0xFFFFu) | (32u << 16)); // dim1 hi | tile_dim0=32
  g1[4] = 128;                                         // tile_dim1=128, tile_dim2=0
  g1[5] = lda;                                         // tensor_dim0_stride (elements)
  g1[6] = 0;
  g1[7] = 0;
  i32x4 z4 = {};
  i32x8 z8 = {};
  // 6-arg form on this toolchain: (g0, g1, g2, g3, g4, cpol)
  __builtin_amdgcn_tensor_load_to_lds(g0, g1, z4, z4, z8, 0);
}
static __device__ __forceinline__ unsigned lds_addr_of(const void* p) {
  return (unsigned)(unsigned long long)p;              // low 32 bits = LDS offset
}
#endif

// ================= embedding + complex positional encoding =================
__global__ __launch_bounds__(256) void embed_kernel(
    const int* __restrict__ data, const float* __restrict__ emb,
    float* __restrict__ xr, float* __restrict__ xi) {
  const int bt = blockIdx.x;
  const int t  = bt & (TSEQ - 1);
  #pragma unroll
  for (int i = 0; i < 4; ++i) {
    int d = threadIdx.x + i * 256;
    if (d >= DM) break;
    int f = d >> 7, c = d & 127;
    int tok = data[(size_t)bt * 7 + f];
    float er = emb[((size_t)(0 + f) * 361 + tok) * 128 + c];
    float ei = emb[((size_t)(7 + f) * 361 + tok) * 128 + c];
    int j = d >> 1;
    float ang = (float)t * __expf((float)(2 * j) * (-9.210340371976184f / (float)DM));
    xr[(size_t)bt * DM + d] = er + __cosf(ang);
    xi[(size_t)bt * DM + d] = ei + __sinf(ang);
  }
}

// ========== complex layernorm (one block per (b,t)); f16 output ==========
__global__ __launch_bounds__(256) void cln_kernel(
    const float* __restrict__ xr, const float* __restrict__ xi,
    const float* __restrict__ p,  // [4,DM]: gr, gi, br, bi
    _Float16* __restrict__ outr, _Float16* __restrict__ outi) {
  const int bt = blockIdx.x;
  const float* xrp = xr + (size_t)bt * DM;
  const float* xip = xi + (size_t)bt * DM;
  float sr = 0.f, si = 0.f, sm = 0.f, sm2 = 0.f;
  for (int d = threadIdx.x; d < DM; d += 256) {
    float a = xrp[d], b = xip[d];
    float mg = fast_sqrt(a * a + b * b + EPSV);
    sr += a; si += b; sm += mg; sm2 += mg * mg;
  }
  __shared__ float red[256];
  auto reduce = [&](float v) -> float {
    red[threadIdx.x] = v; __syncthreads();
    for (int s = 128; s > 0; s >>= 1) {
      if ((int)threadIdx.x < s) red[threadIdx.x] += red[threadIdx.x + s];
      __syncthreads();
    }
    float r = red[0]; __syncthreads(); return r;
  };
  float Sr = reduce(sr), Si = reduce(si), Sm = reduce(sm), Sm2 = reduce(sm2);
  float mur = Sr / DM, mui = Si / DM, mm = Sm / DM;
  float var = (Sm2 - (float)DM * mm * mm) / (float)(DM - 1);
  float stdv = fast_sqrt(fmaxf(var, 0.f)) + EPSV;
  float inv = fast_rcp(stdv);
  for (int d = threadIdx.x; d < DM; d += 256) {
    float a = (xrp[d] - mur) * inv;
    float b = (xip[d] - mui) * inv;
    float gr = p[d], gi = p[DM + d], br = p[2 * DM + d], bi = p[3 * DM + d];
    outr[(size_t)bt * DM + d] = (_Float16)(gr * a - gi * b + br);
    outi[(size_t)bt * DM + d] = (_Float16)(gr * b + gi * a + bi);
  }
}

// ================= generic complex GEMM via WMMA f16 =================
//   Cr = Ar*Wr - Ai*Wi + (br - bi)    Ci = Ai*Wr + Ar*Wi + (br + bi)
// A: f16 [M,K] (staged by TDM when available); W: f32 [K,N] converted at staging.
// FLAGS: 1 = residual accumulate (fp32 C += ...), 2 = f16 output, 4 = imag-only
// C index = cbase + (m/Mseg)*segStride + (m%Mseg)*rowStride + n*colStride
#define BM 128
#define BN 128
#define BK 32

template <int FLAGS>
__global__ __launch_bounds__(256) void cgemm_wmma(
    const _Float16* __restrict__ Ar, const _Float16* __restrict__ Ai, int lda,
    const float* __restrict__ Wr, const float* __restrict__ Wi, int ldw,
    const float* __restrict__ br, const float* __restrict__ bi,
    void* Cr, void* Ci,
    int cbase, int Mseg, int segStride, int rowStride, int colStride,
    int N, int K) {
  __shared__ __align__(32) _Float16 sAr[2][BM * BK];   // double-buffered (TDM)
  __shared__ __align__(32) _Float16 sAi[2][BM * BK];
  __shared__ __align__(32) _Float16 sBr[BN * BK];
  __shared__ __align__(32) _Float16 sBi[BN * BK];

  constexpr bool realOnly = (FLAGS & 4) != 0;
  const int tid = threadIdx.x;
  const int lane = tid & 31, wave = tid >> 5;
  const int wm = wave & 3, wn = wave >> 2;         // 4 waves in M, 2 in N
  const int hh = lane >> 4, l16 = lane & 15;
  const int m0 = blockIdx.x * BM;
  const int n0 = blockIdx.y * BN;

  // W staging decomposition
  const int cc0  = (tid >> 4) * 2;                    // k-row pair
  const int nn0  = (tid & 15) * 8;                    // 8 n-values
  const bool fastB = (n0 + BN <= N) && ((ldw & 3) == 0);
  float wv[2][2][8];                                  // [stream][krow][n]

  auto loadW = [&](int k0) {
    #pragma unroll
    for (int s = 0; s < 2; ++s) {
      const float* Wp = s ? Wi : Wr;
      #pragma unroll
      for (int kr = 0; kr < 2; ++kr) {
        const float* rowp = Wp + (size_t)(k0 + cc0 + kr) * ldw + n0 + nn0;
        if (fastB) {
          #pragma unroll
          for (int j4 = 0; j4 < 2; ++j4) {
            v4f v = *(const v4f*)(rowp + j4 * 4);
            #pragma unroll
            for (int e = 0; e < 4; ++e) wv[s][kr][j4 * 4 + e] = v[e];
          }
        } else {
          #pragma unroll
          for (int jj = 0; jj < 8; ++jj)
            wv[s][kr][jj] = (n0 + nn0 + jj < N) ? rowp[jj] : 0.f;
        }
      }
    }
  };
  auto storeW = [&]() {
    #pragma unroll
    for (int jj = 0; jj < 8; ++jj) {
      H2 p0, p1;
      p0.h[0] = (_Float16)wv[0][0][jj]; p0.h[1] = (_Float16)wv[0][1][jj];
      p1.h[0] = (_Float16)wv[1][0][jj]; p1.h[1] = (_Float16)wv[1][1][jj];
      *(unsigned int*)&sBr[(nn0 + jj) * BK + cc0] = p0.u;
      *(unsigned int*)&sBi[(nn0 + jj) * BK + cc0] = p1.u;
    }
  };

#if HAVE_TDM
  auto issueA = [&](int k0, int par) {
    if (wave == 0) {   // one TDM issue per stream per tile (whole-tile DMA)
      tdm_load_a(Ar + (size_t)m0 * lda + k0, lds_addr_of(&sAr[par][0]), lda);
      tdm_load_a(Ai + (size_t)m0 * lda + k0, lds_addr_of(&sAi[par][0]), lda);
    }
  };
#else
  const int arow = tid >> 1, acol = (tid & 1) * 16;   // A: 1 v16h per thread
  v16h rA0, rA1;
  auto loadA = [&](int k0) {
    rA0 = *(const v16h*)(Ar + (size_t)(m0 + arow) * lda + k0 + acol);
    rA1 = *(const v16h*)(Ai + (size_t)(m0 + arow) * lda + k0 + acol);
  };
  auto storeA = [&](int par) {
    *(v16h*)&sAr[par][arow * BK + acol] = rA0;
    *(v16h*)&sAi[par][arow * BK + acol] = rA1;
  };
#endif

  v8f accr[2][4] = {};
  v8f acci[2][4] = {};

  // ---- prologue: stage tile 0 ----
  loadW(0);
  storeW();
#if HAVE_TDM
  issueA(0, 0);
  __builtin_amdgcn_s_wait_tensorcnt(0);
#else
  loadA(0);
  storeA(0);
#endif
  __syncthreads();

  int par = 0;
  for (int k0 = 0; k0 < K; k0 += BK) {
    const bool notLast = (k0 + BK < K);
    if (notLast) {
#if HAVE_TDM
      issueA(k0 + BK, par ^ 1);   // async DMA overlaps with compute below
#else
      loadA(k0 + BK);
#endif
      loadW(k0 + BK);
    }

    // ---- compute current tile ----
    V16 fa[2], fb[2], fna[2];
    #pragma unroll
    for (int tm = 0; tm < 2; ++tm) {
      const _Float16* pa = &sAr[par][(wm * 32 + tm * 16 + l16) * BK];
      const _Float16* pi = &sAi[par][(wm * 32 + tm * 16 + l16) * BK];
      fa[tm].h[0] = *(const v8h*)(pa + hh * 8);
      fa[tm].h[1] = *(const v8h*)(pa + 16 + hh * 8);
      fb[tm].h[0] = *(const v8h*)(pi + hh * 8);
      fb[tm].h[1] = *(const v8h*)(pi + 16 + hh * 8);
      fna[tm].v = -fb[tm].v;  // -Ai fragment (f16 WMMA NEG bits are C-only)
    }
    V16 gbr[4], gbi[4];
    #pragma unroll
    for (int tn = 0; tn < 4; ++tn) {
      const _Float16* pr = &sBr[(wn * 64 + tn * 16 + l16) * BK + hh * 16];
      const _Float16* pi = &sBi[(wn * 64 + tn * 16 + l16) * BK + hh * 16];
      gbr[tn].h[0] = *(const v8h*)(pr - hh * 8);
      gbr[tn].h[1] = *(const v8h*)(pr - hh * 8 + 16);
      gbi[tn].h[0] = *(const v8h*)(pi - hh * 8);
      gbi[tn].h[1] = *(const v8h*)(pi - hh * 8 + 16);
    }
    #pragma unroll
    for (int tm = 0; tm < 2; ++tm)
      #pragma unroll
      for (int tn = 0; tn < 4; ++tn) {
        if constexpr (!realOnly) {
          accr[tm][tn] = wmma_f16(fa[tm].v,  gbr[tn].v, accr[tm][tn]);
          accr[tm][tn] = wmma_f16(fna[tm].v, gbi[tn].v, accr[tm][tn]);
        }
        acci[tm][tn] = wmma_f16(fb[tm].v, gbr[tn].v, acci[tm][tn]);
        acci[tm][tn] = wmma_f16(fa[tm].v, gbi[tn].v, acci[tm][tn]);
      }
    __syncthreads();                 // everyone done reading sB / sA[par]
    if (notLast) {
      storeW();
#if HAVE_TDM
      __builtin_amdgcn_s_wait_tensorcnt(0);  // next A tile landed (nop for waves 1..7)
#else
      storeA(par ^ 1);
#endif
      __syncthreads();
    }
    par ^= 1;
  }

  // ---- epilogue (C layout: VGPR r -> row r + 8*hh, col l16) ----
  #pragma unroll
  for (int tm = 0; tm < 2; ++tm)
    #pragma unroll
    for (int tn = 0; tn < 4; ++tn)
      #pragma unroll
      for (int r = 0; r < 8; ++r) {
        int m = m0 + wm * 32 + tm * 16 + r + hh * 8;
        int n = n0 + wn * 64 + tn * 16 + l16;
        if (n >= N) continue;
        size_t cidx = (size_t)cbase + (size_t)(m / Mseg) * segStride +
                      (size_t)(m % Mseg) * rowStride + (size_t)n * colStride;
        float vb_r = br ? br[n] : 0.f;
        float vb_i = bi ? bi[n] : 0.f;
        float vr = accr[tm][tn][r] + vb_r - vb_i;
        float vi = acci[tm][tn][r] + vb_r + vb_i;
        if constexpr ((FLAGS & 2) != 0) {
          if constexpr (!realOnly) ((_Float16*)Cr)[cidx] = (_Float16)vr;
          ((_Float16*)Ci)[cidx] = (_Float16)vi;
        } else {
          float* cr = (float*)Cr; float* ci = (float*)Ci;
          if constexpr ((FLAGS & 1) != 0) { vr += cr[cidx]; vi += ci[cidx]; }
          if constexpr (!realOnly) cr[cidx] = vr;
          ci[cidx] = vi;
        }
      }
}

// ================= complex flash attention (1 wave per 16-row block) ========
// Q,K: f16 [B,H,T,HS]; V: f16 TRANSPOSED [B,H,HS,T]; out: f16 [B,T,D]
__global__ __launch_bounds__(32) void cattn_wmma(
    const _Float16* __restrict__ gqr, const _Float16* __restrict__ gqi,
    const _Float16* __restrict__ gkr, const _Float16* __restrict__ gki,
    const _Float16* __restrict__ gvr, const _Float16* __restrict__ gvi,
    _Float16* __restrict__ gor, _Float16* __restrict__ goi) {
  __shared__ __align__(32) _Float16 Pl[16 * 32];
  const int lane = threadIdx.x;
  const int hh = lane >> 4, l16 = lane & 15;
  const int rb = blockIdx.x, h = blockIdx.y, b = blockIdx.z;
  const int t0 = rb * 16;
  const size_t bh = ((size_t)b * NH + h) * TSEQ * HSZ;
  const _Float16* Qr = gqr + bh; const _Float16* Qi = gqi + bh;
  const _Float16* Kr = gkr + bh; const _Float16* Ki = gki + bh;
  const _Float16* Vr = gvr + bh; const _Float16* Vi = gvi + bh;  // [HS,T]
  const float scale = 0.09449111825230679f;  // HS^-0.5

  // ---- Q fragments: 4 k-steps of 32 over HS=112 (last 16 zero-padded) ----
  V16 aqr[4], aqi[4], nqr[4];
  #pragma unroll
  for (int s = 0; s < 4; ++s) {
    const _Float16* pr = Qr + (size_t)(t0 + l16) * HSZ + s * 32;
    const _Float16* pi = Qi + (size_t)(t0 + l16) * HSZ + s * 32;
    aqr[s].h[0] = *(const v8h*)(pr + hh * 8);
    aqi[s].h[0] = *(const v8h*)(pi + hh * 8);
    if (s < 3) {
      aqr[s].h[1] = *(const v8h*)(pr + 16 + hh * 8);
      aqi[s].h[1] = *(const v8h*)(pi + 16 + hh * 8);
    } else {
      v8h z = {}; aqr[s].h[1] = z; aqi[s].h[1] = z;
    }
    nqr[s].v = -aqr[s].v;
  }

  v8f o_r[7] = {}, o_i[7] = {};
  float mrow[8], lrow[8];
  #pragma unroll
  for (int r = 0; r < 8; ++r) { mrow[r] = -1e30f; lrow[r] = 0.f; }

  const int ncb = (t0 + 15) / 32 + 1;
  for (int cb = 0; cb < ncb; ++cb) {
    const int u0 = cb * 32;
    v8f wm2[2];
    #pragma unroll
    for (int ut = 0; ut < 2; ++ut) {
      v8f sre = {}, sim = {};
      const int urow = u0 + ut * 16 + l16;
      const _Float16* pkr = Kr + (size_t)urow * HSZ;
      const _Float16* pki = Ki + (size_t)urow * HSZ;
      #pragma unroll
      for (int s = 0; s < 4; ++s) {
        V16 bkr, bki;
        bkr.h[0] = *(const v8h*)(pkr + s * 32 + hh * 8);
        bki.h[0] = *(const v8h*)(pki + s * 32 + hh * 8);
        if (s == 3) {
          v8h z = {}; bkr.h[1] = z; bki.h[1] = z;
        } else {
          bkr.h[1] = *(const v8h*)(pkr + s * 32 + 16 + hh * 8);
          bki.h[1] = *(const v8h*)(pki + s * 32 + 16 + hh * 8);
        }
        sre = wmma_f16(aqr[s].v, bkr.v, sre);   //  qr.kr
        sre = wmma_f16(aqi[s].v, bki.v, sre);   // +qi.ki
        sim = wmma_f16(aqi[s].v, bkr.v, sim);   //  qi.kr
        sim = wmma_f16(nqr[s].v, bki.v, sim);   // -qr.ki
      }
      #pragma unroll
      for (int r = 0; r < 8; ++r) {
        float re = sre[r], im = sim[r];
        float mg = fast_sqrt(re * re + im * im + EPSV) * scale;
        int u = u0 + ut * 16 + l16;
        int t = t0 + r + 8 * hh;
        wm2[ut][r] = (u <= t) ? mg : -1e30f;
      }
    }
    // ---- online softmax across this 32-column block ----
    float pa[8], pb[8], alpha[8];
    #pragma unroll
    for (int r = 0; r < 8; ++r) {
      float tmax = fmaxf(wm2[0][r], wm2[1][r]);
      #pragma unroll
      for (int off = 1; off < 16; off <<= 1)
        tmax = fmaxf(tmax, __shfl_xor(tmax, off, 32));
      float mnew = fmaxf(mrow[r], tmax);
      alpha[r] = __expf(mrow[r] - mnew);
      mrow[r] = mnew;
      pa[r] = __expf(wm2[0][r] - mnew);
      pb[r] = __expf(wm2[1][r] - mnew);
      float rs = pa[r] + pb[r];
      #pragma unroll
      for (int off = 1; off < 16; off <<= 1)
        rs += __shfl_xor(rs, off, 32);
      lrow[r] = lrow[r] * alpha[r] + rs;
      Pl[(r + 8 * hh) * 32 + l16]      = (_Float16)pa[r];
      Pl[(r + 8 * hh) * 32 + 16 + l16] = (_Float16)pb[r];
    }
    #pragma unroll
    for (int vt = 0; vt < 7; ++vt)
      #pragma unroll
      for (int r = 0; r < 8; ++r) { o_r[vt][r] *= alpha[r]; o_i[vt][r] *= alpha[r]; }
    __syncthreads();
    V16 pf;
    { const _Float16* pp = &Pl[l16 * 32];
      pf.h[0] = *(const v8h*)(pp + hh * 8);
      pf.h[1] = *(const v8h*)(pp + 16 + hh * 8); }
    __syncthreads();
    // ---- O += P * V  (V transposed: B-fragment loads are contiguous) ----
    #pragma unroll
    for (int vt = 0; vt < 7; ++vt) {
      V16 bvr, bvi;
      const size_t vrow = (size_t)(vt * 16 + l16) * TSEQ + u0 + hh * 8;
      bvr.h[0] = *(const v8h*)(Vr + vrow);
      bvr.h[1] = *(const v8h*)(Vr + vrow + 16);
      bvi.h[0] = *(const v8h*)(Vi + vrow);
      bvi.h[1] = *(const v8h*)(Vi + vrow + 16);
      o_r[vt] = wmma_f16(pf.v, bvr.v, o_r[vt]);
      o_i[vt] = wmma_f16(pf.v, bvi.v, o_i[vt]);
    }
  }
  // ---- normalize, write head-concat output [B,T,D] (f16) ----
  #pragma unroll
  for (int r = 0; r < 8; ++r) {
    float inv = fast_rcp(lrow[r]);
    int t = t0 + r + 8 * hh;
    size_t base = ((size_t)b * TSEQ + t) * DM + h * HSZ;
    #pragma unroll
    for (int vt = 0; vt < 7; ++vt) {
      gor[base + vt * 16 + l16] = (_Float16)(o_r[vt][r] * inv);
      goi[base + vt * 16 + l16] = (_Float16)(o_i[vt][r] * inv);
    }
  }
}

// ================= host orchestration =================
extern "C" void kernel_launch(void* const* d_in, const int* in_sizes, int n_in,
                              void* d_out, int out_size, void* d_ws, size_t ws_size,
                              hipStream_t stream) {
  (void)in_sizes; (void)n_in; (void)out_size; (void)ws_size;
  const int*   data  = (const int*)  d_in[0];
  const float* emb   = (const float*)d_in[1];
  const float* ln1   = (const float*)d_in[2];
  const float* Wqkv  = (const float*)d_in[3];
  const float* Wproj = (const float*)d_in[4];
  const float* bproj = (const float*)d_in[5];
  const float* ln2   = (const float*)d_in[6];
  const float* W1    = (const float*)d_in[7];
  const float* b1    = (const float*)d_in[8];
  const float* W2    = (const float*)d_in[9];
  const float* b2    = (const float*)d_in[10];
  const float* lnf   = (const float*)d_in[11];
  const float* Whead[4] = {(const float*)d_in[12], (const float*)d_in[13],
                           (const float*)d_in[14], (const float*)d_in[15]};
  const int NhArr[4] = {181, 361, 151, 92};

  // workspace layout (float units); total 8*BTD floats (~117 MB)
  float* ws = (float*)d_ws;
  float* XR = ws;               float* XI = ws + BTD;            // f32 residual
  _Float16* HRh = (_Float16*)(ws + 2 * BTD); _Float16* HIh = HRh + BTD;
  _Float16* ARh = (_Float16*)(ws + 3 * BTD); _Float16* AIh = ARh + BTD;
  _Float16* Qh  = (_Float16*)(ws + 4 * BTD);   // 6 f16 qkv buffers
  _Float16* QRh = Qh + 0 * BTD; _Float16* QIh = Qh + 1 * BTD;
  _Float16* KRh = Qh + 2 * BTD; _Float16* KIh = Qh + 3 * BTD;
  _Float16* VtR = Qh + 4 * BTD; _Float16* VtI = Qh + 5 * BTD;    // [B,H,HS,T]
  _Float16* FRh = (_Float16*)(ws + 4 * BTD);   // overlaps QKV (dead at FFN time)
  _Float16* FIh = FRh + (size_t)BT * DFF;

  embed_kernel<<<dim3(BT), dim3(256), 0, stream>>>(data, emb, XR, XI);

  for (int l = 0; l < 4; ++l) {
    cln_kernel<<<dim3(BT), dim3(256), 0, stream>>>(XR, XI, ln1 + (size_t)l * 4 * DM, HRh, HIh);
    _Float16* outR[3] = {KRh, QRh, VtR};   // reference order: k, q, v
    _Float16* outI[3] = {KIh, QIh, VtI};
    for (int p = 0; p < 3; ++p)
      for (int h = 0; h < NH; ++h) {
        const float* w0 = Wqkv + ((((size_t)l * 3 + p) * 2 + 0) * NH + h) * (size_t)DM * HSZ;
        const float* w1 = Wqkv + ((((size_t)l * 3 + p) * 2 + 1) * NH + h) * (size_t)DM * HSZ;
        const int rowS = (p == 2) ? 1 : HSZ;      // V written transposed [HS,T]
        const int colS = (p == 2) ? TSEQ : 1;
        cgemm_wmma<2><<<dim3(BT / BM, 1), dim3(256), 0, stream>>>(
            HRh, HIh, DM, w0, w1, HSZ, nullptr, nullptr,
            (void*)outR[p], (void*)outI[p],
            h * TSEQ * HSZ, TSEQ, NH * TSEQ * HSZ, rowS, colS, HSZ, DM);
      }
    cattn_wmma<<<dim3(TSEQ / 16, NH, BB), dim3(32), 0, stream>>>(
        QRh, QIh, KRh, KIh, VtR, VtI, ARh, AIh);
    // output projection with residual into XR/XI
    cgemm_wmma<1><<<dim3(BT / BM, DM / BN), dim3(256), 0, stream>>>(
        ARh, AIh, DM,
        Wproj + ((size_t)l * 2 + 0) * DM * DM, Wproj + ((size_t)l * 2 + 1) * DM * DM, DM,
        bproj + ((size_t)l * 2 + 0) * DM, bproj + ((size_t)l * 2 + 1) * DM,
        XR, XI, 0, BT, 0, DM, 1, DM, DM);
    cln_kernel<<<dim3(BT), dim3(256), 0, stream>>>(XR, XI, ln2 + (size_t)l * 4 * DM, HRh, HIh);
    cgemm_wmma<2><<<dim3(BT / BM, DFF / BN), dim3(256), 0, stream>>>(
        HRh, HIh, DM,
        W1 + ((size_t)l * 2 + 0) * DM * DFF, W1 + ((size_t)l * 2 + 1) * DM * DFF, DFF,
        b1 + ((size_t)l * 2 + 0) * DFF, b1 + ((size_t)l * 2 + 1) * DFF,
        (void*)FRh, (void*)FIh, 0, BT, 0, DFF, 1, DFF, DM);
    // modReLU: m = sqrt(fr^2+fi^2+1e-4) > 0 always => relu(m)/m == 1.0 exactly
    // -> identity; no kernel needed.
    cgemm_wmma<1><<<dim3(BT / BM, DM / BN), dim3(256), 0, stream>>>(
        FRh, FIh, DFF,
        W2 + ((size_t)l * 2 + 0) * DFF * DM, W2 + ((size_t)l * 2 + 1) * DFF * DM, DM,
        b2 + ((size_t)l * 2 + 0) * DM, b2 + ((size_t)l * 2 + 1) * DM,
        XR, XI, 0, BT, 0, DM, 1, DM, DFF);
  }
  cln_kernel<<<dim3(BT), dim3(256), 0, stream>>>(XR, XI, lnf, HRh, HIh);

  // heads: out = yr @ W[:D] + yi @ W[D:]  (imag-only mode: Ci = Ai*Wr + Ar*Wi)
  float* out = (float*)d_out;
  size_t off = 0;
  for (int hd = 0; hd < 4; ++hd) {
    int Nh = NhArr[hd];
    cgemm_wmma<4><<<dim3(BT / BM, (Nh + BN - 1) / BN), dim3(256), 0, stream>>>(
        HRh, HIh, DM,
        Whead[hd] + (size_t)DM * Nh /*Wr = bottom half*/, Whead[hd] /*Wi = top half*/, Nh,
        nullptr, nullptr, nullptr, (void*)(out + off),
        0, BT, 0, Nh, 1, Nh, DM);
    off += (size_t)BT * Nh;
  }
}
